// RGCNStack_9998683865852
// MI455X (gfx1250) — matrix operand order repaired and checked by reading, not compiled, not run on hardware
//
#include <hip/hip_runtime.h>
#include <stdint.h>

// Problem constants (from reference)
#define NNODES 20000
#define NREL   48
#define NBASE  12
#define NEDGE  640000
#define F      128          // all layer widths are 128
#define FD     64           // F/2 dwords per bf16 row
#define LDSPITCH 68         // dword pitch for 16x128 bf16 A tile (bank-conflict free)

typedef float  v8f   __attribute__((ext_vector_type(8)));
typedef __bf16 v16bf __attribute__((ext_vector_type(16)));
typedef __bf16 bf2   __attribute__((ext_vector_type(2)));

union Frag { uint4 q[2]; v16bf v; };
union QU   { uint4 q; bf2 b[4]; };

// pack two floats to bf16 pair via hardware cvt (fptrunc)
__device__ __forceinline__ uint32_t cvt2(float a, float b) {
    union { bf2 v; uint32_t u; } u;
    u.v = bf2{ (__bf16)a, (__bf16)b };
    return u.u;
}

// ---------------------------------------------------------------------------
// Wt[r][o][k] = sum_b comp[r,b] * basis[b,k,o]   (bf16, transposed for B-frags)
// slot r == NREL holds root^T.  One dword (2 k-values) per thread.
// ---------------------------------------------------------------------------
__global__ void wt_kernel(const float* __restrict__ comp,
                          const float* __restrict__ basis,
                          const float* __restrict__ root,
                          uint32_t* __restrict__ wt) {
    int t = blockIdx.x * 256 + threadIdx.x;            // (NREL+1)*128*64 threads
    int r   = t >> 13;            // /8192
    int rem = t & 8191;
    int o   = rem >> 6;
    int k0  = (rem & 63) << 1;
    float a, b;
    if (r < NREL) {
        a = 0.f; b = 0.f;
        #pragma unroll
        for (int bb = 0; bb < NBASE; ++bb) {
            float c = comp[r * NBASE + bb];
            a = fmaf(c, basis[bb * (F * F) + k0 * F + o], a);
            b = fmaf(c, basis[bb * (F * F) + (k0 + 1) * F + o], b);
        }
    } else {
        a = root[k0 * F + o];
        b = root[(k0 + 1) * F + o];
    }
    wt[t] = cvt2(a, b);
}

// ---------------------------------------------------------------------------
// cnt[etype*N + dst] += 1   (computed once, reused by all 3 layers)
// ---------------------------------------------------------------------------
__global__ void cnt_kernel(const int* __restrict__ dst,
                           const int* __restrict__ etype,
                           float* __restrict__ cnt) {
    int e = blockIdx.x * 256 + threadIdx.x;
    if (e < NEDGE) {
        size_t seg = (size_t)etype[e] * NNODES + (size_t)dst[e];
        atomicAdd(&cnt[seg], 1.0f);
    }
}

// cnt[i] <- 1/max(cnt[i],1): hoists the IEEE divide out of the GEMM hot loop
__global__ void inv_kernel(float* __restrict__ cnt) {
    int i = blockIdx.x * 256 + threadIdx.x;
    cnt[i] = 1.0f / fmaxf(cnt[i], 1.0f);
}

// ---------------------------------------------------------------------------
// d_out[:, 384:512] = emb
// ---------------------------------------------------------------------------
__global__ void emb_copy_kernel(const float* __restrict__ emb,
                                float* __restrict__ out) {
    int t = blockIdx.x * 256 + threadIdx.x;            // N*F threads
    int n = t >> 7, f = t & 127;
    out[(size_t)n * 512 + 384 + f] = emb[(size_t)n * F + f];
}

// ---------------------------------------------------------------------------
// Scatter: msg[etype*N+dst][:] += bf16(x[src][:])  via global_atomic_pk_add_bf16
// One wave (32 lanes) per edge; each lane covers 4 features (2 packed atomics).
// ---------------------------------------------------------------------------
__global__ void scatter_kernel(const float* __restrict__ x, int ldx,
                               const int* __restrict__ src,
                               const int* __restrict__ dst,
                               const int* __restrict__ etype,
                               uint32_t* __restrict__ msg) {
    int t = blockIdx.x * 256 + threadIdx.x;
    int e = t >> 5, lane = t & 31;
    if (e >= NEDGE) return;
    int s = src[e];
    size_t seg = (size_t)etype[e] * NNODES + (size_t)dst[e];
    const float* xr = x + (size_t)s * ldx + (lane << 2);
    float4 f = *(const float4*)xr;
    uint32_t p0 = cvt2(f.x, f.y);
    uint32_t p1 = cvt2(f.z, f.w);
    unsigned long long addr =
        (unsigned long long)(uintptr_t)msg + ((seg << 6) + (lane << 1)) * 4ull;
    asm volatile("global_atomic_pk_add_bf16 %0, %1, off" :: "v"(addr), "v"(p0) : "memory");
    asm volatile("global_atomic_pk_add_bf16 %0, %1, off" :: "v"(addr + 4ull), "v"(p1) : "memory");
}

// ---------------------------------------------------------------------------
// GEMM: out[n, off+o] = relu( sum_{r<=R} A_r[n,:] @ Wt[r][o,:]^T + bias[o] )
//   A_r = msg[r]*cntinv for r<R (bf16, scaled with v_pk_mul_bf16),
//   A_R = bf16(x) (root term).
// 16-node tile per WG, 8 waves x 16-col tiles, K=128 = 4 x WMMA(16x16x32 bf16).
// Double-buffered LDS A tile: one barrier per relation, next relation's global
// loads issue before the current relation's WMMAs.
// ---------------------------------------------------------------------------
__global__ __launch_bounds__(256) void gemm_kernel(
        const uint32_t* __restrict__ msg, const float* __restrict__ cntinv,
        const float* __restrict__ x, int ldx,
        const uint32_t* __restrict__ wt, const float* __restrict__ bias,
        float* __restrict__ out, int out_off) {
    __shared__ uint32_t lds_a[2][16 * LDSPITCH];

    const int n0   = blockIdx.x * 16;
    const int tid  = threadIdx.x;
    const int wave = tid >> 5;
    const int lane = tid & 31;
    const int m16  = lane & 15;
    const int sel  = lane >> 4;                 // 0: lanes 0-15, 1: lanes 16-31
    const int ocol = (wave << 4) + m16;         // this lane's output column

    // LDS-fill assignment: 16 threads per row, 4 dwords (8 bf16) each
    const int frow = tid >> 4;
    const int fcol = (tid & 15) << 2;

    v8f acc = {};
    int cur = 0;

    // ---- prologue: stage relation 0 into buffer 0 ----
    {
        size_t seg = (size_t)(n0 + frow);
        QU u;
        u.q = *(const uint4*)(msg + (seg << 6) + fcol);
        __bf16 ib = (__bf16)cntinv[seg];
        bf2 iv = { ib, ib };
        u.b[0] *= iv; u.b[1] *= iv; u.b[2] *= iv; u.b[3] *= iv;
        *(uint4*)(&lds_a[0][frow * LDSPITCH + fcol]) = u.q;
    }
    __syncthreads();

    for (int r = 0; r <= NREL; ++r) {
        // ---- issue global loads for tile r+1 (hidden under WMMAs) ----
        const bool fill_msg  = (r + 1 < NREL);
        const bool fill_root = (r + 1 == NREL);
        QU u;
        bf2 iv = {};
        if (fill_msg) {
            size_t seg = (size_t)(r + 1) * NNODES + (size_t)(n0 + frow);
            const uint32_t* mrow = msg + (seg << 6);
            u.q = *(const uint4*)(mrow + fcol);
            __bf16 ib = (__bf16)cntinv[seg];
            iv = bf2{ ib, ib };
            if (r + 2 < NREL)   // global_prefetch_b8 of the relation after that
                __builtin_prefetch((const void*)(mrow + ((size_t)NNODES << 6) + fcol), 0, 1);
        } else if (fill_root) {
            const float* xr = x + (size_t)(n0 + frow) * ldx + (fcol << 1);
            float4 a = *(const float4*)xr;
            float4 b = *(const float4*)(xr + 4);
            u.q.x = cvt2(a.x, a.y); u.q.y = cvt2(a.z, a.w);
            u.q.z = cvt2(b.x, b.y); u.q.w = cvt2(b.z, b.w);
        }

        // ---- 4 WMMAs on current buffer ----
        const uint32_t* br = wt + (((size_t)r * F + ocol) << 6) + (sel << 2);
        const uint32_t* ar = &lds_a[cur][m16 * LDSPITCH + (sel << 2)];
        #pragma unroll
        for (int ks = 0; ks < 4; ++ks) {
            const int K0h = ks << 4;            // K0/2 dwords
            Frag A, B;
            A.q[0] = *(const uint4*)(ar + K0h);
            A.q[1] = *(const uint4*)(ar + K0h + 8);
            B.q[0] = *(const uint4*)(br + K0h);
            B.q[1] = *(const uint4*)(br + K0h + 8);
            acc = __builtin_amdgcn_wmma_f32_16x16x32_bf16(
                      false, A.v, false, B.v, (short)0, acc, false, false);
        }

        // ---- scale (v_pk_mul_bf16) + store tile r+1 into the other buffer ----
        if (fill_msg) {
            u.b[0] *= iv; u.b[1] *= iv; u.b[2] *= iv; u.b[3] *= iv;
            *(uint4*)(&lds_a[cur ^ 1][frow * LDSPITCH + fcol]) = u.q;
        } else if (fill_root) {
            *(uint4*)(&lds_a[cur ^ 1][frow * LDSPITCH + fcol]) = u.q;
        }
        if (r < NREL) __syncthreads();
        cur ^= 1;
    }

    // ---- bias + relu + store to concat slice ----
    const float bv = bias[ocol];
    #pragma unroll
    for (int j = 0; j < 8; ++j) {
        int mm = j + (sel << 3);                // C/D layout: VGPR j, lanes>=16 -> M+8
        float v = fmaxf(acc[j] + bv, 0.0f);
        out[(size_t)(n0 + mm) * 512 + out_off + ocol] = v;
    }
}

// ---------------------------------------------------------------------------
extern "C" void kernel_launch(void* const* d_in, const int* in_sizes, int n_in,
                              void* d_out, int out_size, void* d_ws, size_t ws_size,
                              hipStream_t stream) {
    (void)in_sizes; (void)n_in; (void)out_size; (void)ws_size;
    const float* emb  = (const float*)d_in[0];
    const int* esrc   = (const int*)d_in[1];
    const int* edst   = (const int*)d_in[2];
    const int* etyp   = (const int*)d_in[3];
    const float* comp[3]  = { (const float*)d_in[4],  (const float*)d_in[8],  (const float*)d_in[12] };
    const float* basis[3] = { (const float*)d_in[5],  (const float*)d_in[9],  (const float*)d_in[13] };
    const float* root[3]  = { (const float*)d_in[6],  (const float*)d_in[10], (const float*)d_in[14] };
    const float* bias[3]  = { (const float*)d_in[7],  (const float*)d_in[11], (const float*)d_in[15] };
    float* out = (float*)d_out;

    // workspace layout
    char* p = (char*)d_ws;
    uint32_t* wt  = (uint32_t*)p;                      // (R+1)*128*64 dwords (bf16 pairs)
    p += (size_t)(NREL + 1) * F * FD * 4;
    float* cnt    = (float*)p;                         // R*N floats (counts -> reciprocals)
    p += (size_t)NREL * NNODES * 4;
    uint32_t* msg = (uint32_t*)p;                      // R*N*64 dwords (bf16 pairs)

    // counts + reciprocals: once for all layers
    hipMemsetAsync(cnt, 0, (size_t)NREL * NNODES * 4, stream);
    cnt_kernel<<<(NEDGE + 255) / 256, 256, 0, stream>>>(edst, etyp, cnt);
    inv_kernel<<<(NREL * NNODES) / 256, 256, 0, stream>>>(cnt);

    // emb -> out[:, 384:512]
    emb_copy_kernel<<<(NNODES * F) / 256, 256, 0, stream>>>(emb, out);

    const float* xin[3] = { emb, out + 256, out + 128 };
    const int    ldx[3] = { F, 512, 512 };
    const int    off[3] = { 256, 128, 0 };

    for (int l = 0; l < 3; ++l) {
        wt_kernel<<<((NREL + 1) * F * FD) / 256, 256, 0, stream>>>(
            comp[l], basis[l], root[l], wt);
        hipMemsetAsync(msg, 0, (size_t)NREL * NNODES * F * 2, stream);
        scatter_kernel<<<(NEDGE * 32) / 256, 256, 0, stream>>>(
            xin[l], ldx[l], esrc, edst, etyp, msg);
        gemm_kernel<<<NNODES / 16, 256, 0, stream>>>(
            msg, cnt, xin[l], ldx[l], wt, bias[l], out, off[l]);
    }
}
